// GAT_7937099563689
// MI455X (gfx1250) — compile-verified, matrix operand order checked by reading, not run on hardware
//
#include <hip/hip_runtime.h>
#include <math.h>

typedef __attribute__((ext_vector_type(16))) _Float16 v16h;
typedef __attribute__((ext_vector_type(8)))  _Float16 v8h;
typedef __attribute__((ext_vector_type(8)))  float    v8f;

#define NEG_SLOPE 0.2f

// ---------------------------------------------------------------------------
// Generic fill
// ---------------------------------------------------------------------------
__global__ void fill_kernel(float* __restrict__ p, int n, float v) {
    int i = blockIdx.x * blockDim.x + threadIdx.x;
    if (i < n) p[i] = v;
}

// ---------------------------------------------------------------------------
// WMMA GEMM: C[M,NcReal] = A[M,128] * B[128,NcReal], fp32 in/out, f16 multiply,
// f32 accumulate.  K fixed at 128 (both GAT layers).
//
// Block = 256 threads = 8 waves.  The block stages the full B panel
// (NT*16 cols x 128 K) into LDS as f16 ([col][K] rows padded to 136 halves ->
// 16B aligned).  Each wave computes a 16-row x NT*16-col strip.  The lane's
// entire A row (4 x v16h fragments) is preloaded+converted to f16 in VGPRs
// before the k-loop (16 batched global_load_b128, single wait), so the
// steady-state loop is pure ds_load_b128 + v_wmma.  4*NT static WMMAs.
// grid.x = ceil(M/128)
// ---------------------------------------------------------------------------
template<int NT>
__global__ void __launch_bounds__(256) gemm_wmma_kernel(
    const float* __restrict__ A, const float* __restrict__ B,
    float* __restrict__ C, int M, int NcReal)
{
    constexpr int KK  = 128;
    constexpr int LDB = KK + 8;                    // 136 halves = 272B, 16B aligned
    __shared__ __align__(16) _Float16 Bs[NT * 16][LDB];

    const int tid = threadIdx.x;

    // ---- stage B panel: global [k][NcReal] -> Bs[col][k] (f16, zero-padded)
    for (int idx = tid; idx < NT * 16 * KK; idx += 256) {
        const int k   = idx / (NT * 16);           // consecutive tid -> consecutive col
        const int col = idx - k * (NT * 16);       //   => coalesced global reads
        const float v = (col < NcReal) ? B[(size_t)k * NcReal + col] : 0.0f;
        Bs[col][k] = (_Float16)v;
    }

    const int wave = tid >> 5;                     // 0..7
    const int lane = tid & 31;
    const int half = lane >> 4;                    // lane group (ISA A/B/D layout)
    const int l    = lane & 15;

    int rowA = blockIdx.x * 128 + wave * 16 + l;
    const int rowStore = blockIdx.x * 128 + wave * 16 + half * 8;
    if (rowA >= M) rowA = M - 1;                   // clamp for loads; stores guarded

    // ---- preload the lane's full A row and convert to f16 fragments.
    // Fragment for k0: halves 0..7 = K = k0+half*8 .. +7,
    //                  halves 8..15 = K = k0+16+half*8 .. +7.
    const float4* Arow = (const float4*)(A + (size_t)rowA * KK);
    v16h afrag[4];
    #pragma unroll
    for (int kb = 0; kb < 4; ++kb) {
        const int kLo = kb * 32 + half * 8;
        const int kHi = kLo + 16;
        const float4 a0 = Arow[kLo >> 2];
        const float4 a1 = Arow[(kLo >> 2) + 1];
        const float4 a2 = Arow[kHi >> 2];
        const float4 a3 = Arow[(kHi >> 2) + 1];
        v16h a;
        a[0]  = (_Float16)a0.x; a[1]  = (_Float16)a0.y;
        a[2]  = (_Float16)a0.z; a[3]  = (_Float16)a0.w;
        a[4]  = (_Float16)a1.x; a[5]  = (_Float16)a1.y;
        a[6]  = (_Float16)a1.z; a[7]  = (_Float16)a1.w;
        a[8]  = (_Float16)a2.x; a[9]  = (_Float16)a2.y;
        a[10] = (_Float16)a2.z; a[11] = (_Float16)a2.w;
        a[12] = (_Float16)a3.x; a[13] = (_Float16)a3.y;
        a[14] = (_Float16)a3.z; a[15] = (_Float16)a3.w;
        afrag[kb] = a;
    }

    v8f acc[NT];
    #pragma unroll
    for (int t = 0; t < NT; ++t) acc[t] = (v8f){};

    __syncthreads();

    #pragma unroll
    for (int kb = 0; kb < 4; ++kb) {
        const int kLo = kb * 32 + half * 8;
        const int kHi = kLo + 16;
        #pragma unroll
        for (int t = 0; t < NT; ++t) {
            // B 32x16 fragment, symmetric to A with col in place of row.
            const _Float16* bp = &Bs[t * 16 + l][0];
            const v8h blo = *(const v8h*)(bp + kLo);   // 16B-aligned ds_load_b128
            const v8h bhi = *(const v8h*)(bp + kHi);
            const v16h b = __builtin_shufflevector(
                blo, bhi, 0, 1, 2, 3, 4, 5, 6, 7, 8, 9, 10, 11, 12, 13, 14, 15);
            acc[t] = __builtin_amdgcn_wmma_f32_16x16x32_f16(
                false, afrag[kb], false, b, (short)0, acc[t], false, false);
        }
    }

    // D layout: lane l (group 0) holds (M=r, N=l) in vgpr r; group 1 holds M=8+r.
    #pragma unroll
    for (int t = 0; t < NT; ++t) {
        const int colg = t * 16 + l;
        if (colg < NcReal) {
            #pragma unroll
            for (int r = 0; r < 8; ++r) {
                const int row = rowStore + r;
                if (row < M) C[(size_t)row * NcReal + colg] = acc[t][r];
            }
        }
    }
}

// ---------------------------------------------------------------------------
// Per-(node,head) attention coefficients: alpha_s = <h, a_src>, alpha_d = <h, a_dst>
// ---------------------------------------------------------------------------
__global__ void alpha_kernel(const float* __restrict__ h,
                             const float* __restrict__ a_s,
                             const float* __restrict__ a_d,
                             float* __restrict__ out_s,
                             float* __restrict__ out_d,
                             int N, int H, int D)
{
    int i = blockIdx.x * blockDim.x + threadIdx.x;   // node*H + head
    if (i >= N * H) return;
    const int node = i / H;
    const int head = i - node * H;
    const float* hp = h + ((size_t)node * H + head) * D;
    const float* av = a_s + head * D;
    const float* bv = a_d + head * D;
    float s = 0.f, d = 0.f;
    for (int k = 0; k < D; ++k) { s += hp[k] * av[k]; d += hp[k] * bv[k]; }
    out_s[i] = s;
    out_d[i] = d;
}

// ---------------------------------------------------------------------------
// Float atomic max via sign-split integer atomics (IEEE-754 order preserving).
// ---------------------------------------------------------------------------
__device__ __forceinline__ void atomicMaxF(float* addr, float val) {
    if (val >= 0.0f)
        atomicMax((int*)addr, __float_as_int(val));
    else
        atomicMin((unsigned int*)addr, (unsigned int)__float_as_int(val));
}

__device__ __forceinline__ float leaky(float v) {
    return (v > 0.0f) ? v : NEG_SLOPE * v;
}

// ---------------------------------------------------------------------------
// Segment max over incoming edges: m[dst,head] = max(leaky(as[src]+ad[dst]))
// ---------------------------------------------------------------------------
__global__ void edge_max_kernel(const int* __restrict__ src,
                                const int* __restrict__ dst,
                                const float* __restrict__ as,
                                const float* __restrict__ ad,
                                float* __restrict__ m, int E, int H)
{
    int i = blockIdx.x * blockDim.x + threadIdx.x;   // edge*H + head
    if (i >= E * H) return;
    const int e = i / H;
    const int head = i - e * H;
    const int s = src[e], d = dst[e];
    const float v = leaky(as[s * H + head] + ad[d * H + head]);
    atomicMaxF(&m[d * H + head], v);
}

// ---------------------------------------------------------------------------
// Segment sum of exp(e - m[dst]) into denom[dst,head]
// ---------------------------------------------------------------------------
__global__ void edge_denom_kernel(const int* __restrict__ src,
                                  const int* __restrict__ dst,
                                  const float* __restrict__ as,
                                  const float* __restrict__ ad,
                                  const float* __restrict__ m,
                                  float* __restrict__ denom, int E, int H)
{
    int i = blockIdx.x * blockDim.x + threadIdx.x;
    if (i >= E * H) return;
    const int e = i / H;
    const int head = i - e * H;
    const int s = src[e], d = dst[e];
    const float v = leaky(as[s * H + head] + ad[d * H + head]);
    const float ex = __expf(v - m[d * H + head]);
    atomicAdd(&denom[d * H + head], ex);
}

// ---------------------------------------------------------------------------
// Aggregation: one block per edge; first H lanes compute alpha into LDS,
// then all H*D lanes gather h[src] * alpha and scatter-add to agg[dst].
// ---------------------------------------------------------------------------
__global__ void edge_agg_kernel(const int* __restrict__ src,
                                const int* __restrict__ dst,
                                const float* __restrict__ h,
                                const float* __restrict__ as,
                                const float* __restrict__ ad,
                                const float* __restrict__ m,
                                const float* __restrict__ denom,
                                float* __restrict__ agg,
                                int E, int H, int D)
{
    __shared__ float alpha_sh[8];
    const int e = blockIdx.x;
    const int t = threadIdx.x;
    const int s = src[e], d = dst[e];
    if (t < H) {
        const float v  = leaky(as[s * H + t] + ad[d * H + t]);
        const float ex = __expf(v - m[d * H + t]);
        alpha_sh[t] = ex / (denom[d * H + t] + 1e-16f);
    }
    __syncthreads();
    const int HD = H * D;
    if (t < HD) {
        const float val = h[(size_t)s * HD + t] * alpha_sh[t / D];
        atomicAdd(&agg[(size_t)d * HD + t], val);
    }
}

// ---------------------------------------------------------------------------
// out = agg + bias  (+ optional ELU)
// ---------------------------------------------------------------------------
__global__ void bias_act_kernel(const float* __restrict__ agg,
                                const float* __restrict__ b,
                                float* __restrict__ out,
                                int N, int C, int doElu)
{
    int i = blockIdx.x * blockDim.x + threadIdx.x;
    if (i >= N * C) return;
    float v = agg[i] + b[i % C];
    if (doElu) v = (v > 0.0f) ? v : (__expf(v) - 1.0f);
    out[i] = v;
}

// ---------------------------------------------------------------------------
// Launch
// ---------------------------------------------------------------------------
extern "C" void kernel_launch(void* const* d_in, const int* in_sizes, int n_in,
                              void* d_out, int out_size, void* d_ws, size_t ws_size,
                              hipStream_t stream)
{
    const float* x      = (const float*)d_in[0];
    const int*   ei     = (const int*)  d_in[1];
    const float* W0     = (const float*)d_in[2];
    const float* a_src0 = (const float*)d_in[3];
    const float* a_dst0 = (const float*)d_in[4];
    const float* b0     = (const float*)d_in[5];
    const float* W1     = (const float*)d_in[6];
    const float* a_src1 = (const float*)d_in[7];
    const float* a_dst1 = (const float*)d_in[8];
    const float* b1     = (const float*)d_in[9];
    float* out = (float*)d_out;

    const int N = in_sizes[0] / 128;   // 50000
    const int E = in_sizes[1] / 2;     // 800000
    const int* src = ei;
    const int* dst = ei + E;

    // Workspace carve-up (floats).  Layer-1 buffers alias layer-0 ones that
    // are dead by the time they are written.
    float* ws = (float*)d_ws;
    size_t o = 0;
    float* h0   = ws + o; o += (size_t)N * 128;   // layer-0 features (reused: h1 = first N*40)
    float* as0  = ws + o; o += (size_t)N * 8;     // reused as as1 (N*1)
    float* ad0  = ws + o; o += (size_t)N * 8;     // reused as ad1
    float* m0   = ws + o; o += (size_t)N * 8;     // reused as m1
    float* dn0  = ws + o; o += (size_t)N * 8;     // reused as dn1
    float* agg0 = ws + o; o += (size_t)N * 128;
    float* hid  = ws + o; o += (size_t)N * 128;   // ELU(layer-0 output)

    const int B256 = 256;
    const float NEG_BIG = -3.0e38f;
    const int gRows = (N + 127) / 128;

    // ---------------- Layer 0: H=8, D=16, concat ----------------
    {
        gemm_wmma_kernel<8><<<gRows, 256, 0, stream>>>(x, W0, h0, N, 128);

        alpha_kernel<<<(N * 8 + B256 - 1) / B256, B256, 0, stream>>>(
            h0, a_src0, a_dst0, as0, ad0, N, 8, 16);

        fill_kernel<<<(N * 8 + B256 - 1) / B256, B256, 0, stream>>>(m0,  N * 8, NEG_BIG);
        fill_kernel<<<(N * 8 + B256 - 1) / B256, B256, 0, stream>>>(dn0, N * 8, 0.0f);
        fill_kernel<<<(N * 128 + B256 - 1) / B256, B256, 0, stream>>>(agg0, N * 128, 0.0f);

        edge_max_kernel<<<(E * 8 + B256 - 1) / B256, B256, 0, stream>>>(
            src, dst, as0, ad0, m0, E, 8);
        edge_denom_kernel<<<(E * 8 + B256 - 1) / B256, B256, 0, stream>>>(
            src, dst, as0, ad0, m0, dn0, E, 8);
        edge_agg_kernel<<<E, 128, 0, stream>>>(
            src, dst, h0, as0, ad0, m0, dn0, agg0, E, 8, 16);

        bias_act_kernel<<<(N * 128 + B256 - 1) / B256, B256, 0, stream>>>(
            agg0, b0, hid, N, 128, 1);
    }

    // ---------------- Layer 1: H=1, D=40, mean over 1 head ----------------
    {
        float* h1  = h0;    // N*40 fits in the dead N*128 region
        float* as1 = as0;
        float* ad1 = ad0;
        float* m1  = m0;
        float* dn1 = dn0;

        gemm_wmma_kernel<3><<<gRows, 256, 0, stream>>>(hid, W1, h1, N, 40);

        alpha_kernel<<<(N + B256 - 1) / B256, B256, 0, stream>>>(
            h1, a_src1, a_dst1, as1, ad1, N, 1, 40);

        fill_kernel<<<(N + B256 - 1) / B256, B256, 0, stream>>>(m1,  N, NEG_BIG);
        fill_kernel<<<(N + B256 - 1) / B256, B256, 0, stream>>>(dn1, N, 0.0f);
        fill_kernel<<<(N * 40 + B256 - 1) / B256, B256, 0, stream>>>(out, N * 40, 0.0f);

        edge_max_kernel<<<(E + B256 - 1) / B256, B256, 0, stream>>>(
            src, dst, as1, ad1, m1, E, 1);
        edge_denom_kernel<<<(E + B256 - 1) / B256, B256, 0, stream>>>(
            src, dst, as1, ad1, m1, dn1, E, 1);
        edge_agg_kernel<<<E, 64, 0, stream>>>(
            src, dst, h1, as1, ad1, m1, dn1, out, E, 1, 40);

        bias_act_kernel<<<(N * 40 + B256 - 1) / B256, B256, 0, stream>>>(
            out, b1, out, N, 40, 0);
    }
}
